// RoadrunnerAttention_4776003633520
// MI455X (gfx1250) — compile-verified, matrix operand order checked by reading, not run on hardware
//
#include <hip/hip_runtime.h>
#include <hip/hip_bf16.h>

typedef __attribute__((ext_vector_type(16))) _Float16 v16h;
typedef __attribute__((ext_vector_type(8)))  float    v8f;

#define BS     2
#define SEQ    2048
#define HID    1024
#define NHEAD  16
#define DHEAD  64
#define RNK    256
#define MTOT   (BS * SEQ)      // 4096 total rows
#define ALPHA_MIX 0.6f

// ---------------------------------------------------------------------------
// WMMA helpers (CDNA5 / gfx1250, wave32)
// ---------------------------------------------------------------------------
__device__ __forceinline__ v8f wmma32(v16h a, v16h b, v8f c) {
  return __builtin_amdgcn_wmma_f32_16x16x32_f16(
      /*neg_a=*/false, a, /*neg_b=*/false, b,
      /*c_mod=*/(short)0, c, /*reuse_a=*/false, /*reuse_b=*/false);
}

// A-fragment (16x32, f16, row-major source) — ISA §7.12.2 layout:
// lanes 0-15 : row M = lane,    K = {0..7, 16..23}
// lanes 16-31: row M = lane-16, K = {8..15, 24..31}
// Also serves B-fragments when the source is B^T row-major
// (fragment column n = lane&15, contiguous K per lane).
__device__ __forceinline__ v16h load_frag_rowmajor(const _Float16* p, int ld) {
  const int lane = threadIdx.x & 31;
  const int row  = lane & 15;
  const int hi   = lane >> 4;
  const _Float16* q = p + (size_t)row * ld;
  const int k0 = hi * 8;
  const int k1 = 16 + hi * 8;
  v16h f;
#pragma unroll
  for (int i = 0; i < 8; ++i) f[i] = q[k0 + i];
#pragma unroll
  for (int i = 0; i < 8; ++i) f[8 + i] = q[k1 + i];
  return f;
}

// Prefetch one fragment's cachelines (global_prefetch_b8, WGP scope).
__device__ __forceinline__ void prefetch_frag(const _Float16* p, int ld) {
  const int lane = threadIdx.x & 31;
  const int row  = lane & 15;
  __builtin_prefetch(p + (size_t)row * ld, 0, 3);
}

// ---------------------------------------------------------------------------
// Conversion / transpose prep kernels (f32 -> f16)
// ---------------------------------------------------------------------------
__global__ void cvt_f32_f16(const float* __restrict__ in, _Float16* __restrict__ out, int n) {
  int i = blockIdx.x * blockDim.x + threadIdx.x;
  if (i < n) out[i] = (_Float16)in[i];
}

__global__ void cvtT_f32_f16(const float* __restrict__ in, _Float16* __restrict__ out,
                             int rows, int cols) {
  // in [rows][cols] (row-major) -> out [cols][rows]
  int i = blockIdx.x * blockDim.x + threadIdx.x;
  if (i < rows * cols) {
    int r = i / cols, c = i - r * cols;
    out[(size_t)c * rows + r] = (_Float16)in[i];
  }
}

// ---------------------------------------------------------------------------
// WMMA GEMM: C[M,N] = epilogue( A[M,K] @ B[K,N] )
//   A: f16 row-major (lda); B given as B^T f16 row-major (Bt[n][k], ldb).
//   One wave computes a 32x64 tile: 2 A-frags x 4 B-frags = 8 WMMAs / K-step.
//   Software pipelined: next K-step's fragments are loaded before this step's
//   WMMAs (partial s_wait_loadcnt), and the K-step after that is prefetched
//   into WGP$/L2 via global_prefetch_b8 — a 2-deep pipeline at zero VGPR cost.
//   Epilogue: per-col scale, per-col bias, scalar mix, f32 accumulate,
//   optional f16 store — outMode 1 stores per-head transposed V layout
//   Vt[b][h][d][s] so attention's P@V B-fragments load contiguously.
// ---------------------------------------------------------------------------
__global__ __launch_bounds__(32)
void gemm_wmma(const _Float16* __restrict__ A, int lda,
               const _Float16* __restrict__ Bt, int ldb,
               int N, int K,
               const float* __restrict__ colScale,
               const float* __restrict__ colBias,
               float outScale, int accumulate, int outMode,
               float* __restrict__ outF,
               _Float16* __restrict__ outH) {
  const int m0 = blockIdx.y * 32;
  const int n0 = blockIdx.x * 64;
  const v8f zero = {0.f, 0.f, 0.f, 0.f, 0.f, 0.f, 0.f, 0.f};
  v8f acc[2][4] = {{zero, zero, zero, zero}, {zero, zero, zero, zero}};

  const _Float16* A0 = A + (size_t)(m0 +  0) * lda;
  const _Float16* A1 = A + (size_t)(m0 + 16) * lda;
  const _Float16* Bp[4] = {Bt + (size_t)(n0 +  0) * ldb,
                           Bt + (size_t)(n0 + 16) * ldb,
                           Bt + (size_t)(n0 + 32) * ldb,
                           Bt + (size_t)(n0 + 48) * ldb};

  // prologue: fragments for k = 0
  v16h a0 = load_frag_rowmajor(A0, lda);
  v16h a1 = load_frag_rowmajor(A1, lda);
  v16h b0 = load_frag_rowmajor(Bp[0], ldb);
  v16h b1 = load_frag_rowmajor(Bp[1], ldb);
  v16h b2 = load_frag_rowmajor(Bp[2], ldb);
  v16h b3 = load_frag_rowmajor(Bp[3], ldb);

  for (int k = 32; k < K; k += 32) {
    // stage 1: issue next K-step's loads (retire under the WMMAs below)
    v16h na0 = load_frag_rowmajor(A0 + k, lda);
    v16h na1 = load_frag_rowmajor(A1 + k, lda);
    v16h nb0 = load_frag_rowmajor(Bp[0] + k, ldb);
    v16h nb1 = load_frag_rowmajor(Bp[1] + k, ldb);
    v16h nb2 = load_frag_rowmajor(Bp[2] + k, ldb);
    v16h nb3 = load_frag_rowmajor(Bp[3] + k, ldb);

    // stage 2: prefetch the K-step after that (clamped; duplicate prefetch ok)
    {
      const int kp = (k + 32 < K) ? (k + 32) : k;
      prefetch_frag(A0 + kp, lda);
      prefetch_frag(A1 + kp, lda);
      prefetch_frag(Bp[0] + kp, ldb);
      prefetch_frag(Bp[1] + kp, ldb);
      prefetch_frag(Bp[2] + kp, ldb);
      prefetch_frag(Bp[3] + kp, ldb);
    }

    acc[0][0] = wmma32(a0, b0, acc[0][0]);
    acc[0][1] = wmma32(a0, b1, acc[0][1]);
    acc[0][2] = wmma32(a0, b2, acc[0][2]);
    acc[0][3] = wmma32(a0, b3, acc[0][3]);
    acc[1][0] = wmma32(a1, b0, acc[1][0]);
    acc[1][1] = wmma32(a1, b1, acc[1][1]);
    acc[1][2] = wmma32(a1, b2, acc[1][2]);
    acc[1][3] = wmma32(a1, b3, acc[1][3]);

    a0 = na0; a1 = na1; b0 = nb0; b1 = nb1; b2 = nb2; b3 = nb3;
  }
  // epilogue K-step
  acc[0][0] = wmma32(a0, b0, acc[0][0]);
  acc[0][1] = wmma32(a0, b1, acc[0][1]);
  acc[0][2] = wmma32(a0, b2, acc[0][2]);
  acc[0][3] = wmma32(a0, b3, acc[0][3]);
  acc[1][0] = wmma32(a1, b0, acc[1][0]);
  acc[1][1] = wmma32(a1, b1, acc[1][1]);
  acc[1][2] = wmma32(a1, b2, acc[1][2]);
  acc[1][3] = wmma32(a1, b3, acc[1][3]);

  const int lane = threadIdx.x & 31;
  const int col  = lane & 15;
  const int hi   = lane >> 4;
#pragma unroll
  for (int w = 0; w < 2; ++w) {
#pragma unroll
    for (int t = 0; t < 4; ++t) {
#pragma unroll
      for (int r = 0; r < 8; ++r) {
        const int gr = m0 + 16 * w + r + 8 * hi;
        const int gc = n0 + t * 16 + col;
        float v = acc[w][t][r];
        if (colScale) v *= colScale[gc];
        if (colBias)  v += colBias[gc];
        v *= outScale;
        if (outF) {
          const size_t idx = (size_t)gr * N + gc;
          if (accumulate) outF[idx] += v;
          else            outF[idx]  = v;
        }
        if (outH) {
          size_t idx;
          if (outMode == 1) {
            // per-head transposed V: Vt[b][h][d][s]
            const int bb = gr >> 11;           // / SEQ
            const int ss = gr & (SEQ - 1);
            const int hh = gc >> 6;            // / DHEAD
            const int dd = gc & (DHEAD - 1);
            idx = (((size_t)bb * NHEAD + hh) * DHEAD + dd) * SEQ + ss;
          } else {
            idx = (size_t)gr * N + gc;
          }
          outH[idx] = (_Float16)v;
        }
      }
    }
  }
}

// ---------------------------------------------------------------------------
// Flash attention (causal), WMMA. One wave per (b, head, 16-query tile).
// Q/K: f16 [B*T, H] (head offset h*DHEAD); V: per-head transposed
// Vt[b][h][d][s] so P@V B-fragments are contiguous. Online softmax in
// registers; P staged through LDS to become an A-fragment. All K/V fragment
// loads for a chunk are issued up front (V retires under QK^T + softmax),
// and the next chunk's K/V cachelines are prefetched into WGP$.
// ---------------------------------------------------------------------------
__global__ __launch_bounds__(32)
void flash_attn_wmma(const _Float16* __restrict__ Q,
                     const _Float16* __restrict__ Km,
                     const _Float16* __restrict__ Vt,
                     _Float16* __restrict__ O) {
  __shared__ _Float16 Plds[16 * 32];

  const int qt = blockIdx.x;
  const int h  = blockIdx.y;
  const int b  = blockIdx.z;
  const int qbase = qt * 16;
  const size_t headOff = (size_t)b * SEQ * HID + (size_t)h * DHEAD;      // Q/K/O
  const size_t headD   = ((size_t)b * NHEAD + h) * DHEAD;                // Vt rows

  const int lane = threadIdx.x & 31;
  const int col  = lane & 15;
  const int hi   = lane >> 4;

  const _Float16* qp = Q + headOff + (size_t)qbase * HID;
  v16h qa0 = load_frag_rowmajor(qp, HID);
  v16h qa1 = load_frag_rowmajor(qp + 32, HID);

  float m[8], l[8];
#pragma unroll
  for (int r = 0; r < 8; ++r) { m[r] = -3.0e38f; l[r] = 0.0f; }
  const v8f zero = {0.f, 0.f, 0.f, 0.f, 0.f, 0.f, 0.f, 0.f};
  v8f acc[4] = {zero, zero, zero, zero};

  const float scale = 0.125f;         // 1/sqrt(DHEAD)
  const int kend = qbase + 16;        // causal bound

  for (int kc = 0; kc < kend; kc += 32) {
    // ---- issue ALL loads for this chunk first ----
    v16h kb[2][2], vb[4];
#pragma unroll
    for (int j = 0; j < 2; ++j) {
      const _Float16* kp = Km + headOff + (size_t)(kc + 16 * j) * HID;
      kb[j][0] = load_frag_rowmajor(kp, HID);
      kb[j][1] = load_frag_rowmajor(kp + 32, HID);
    }
#pragma unroll
    for (int t = 0; t < 4; ++t)
      vb[t] = load_frag_rowmajor(Vt + (headD + t * 16) * SEQ + kc, SEQ);

    // ---- prefetch next key chunk into WGP$ while we compute ----
    {
      const int kn = (kc + 32 < kend) ? (kc + 32) : kc;
      prefetch_frag(Km + headOff + (size_t)kn * HID, HID);
      prefetch_frag(Km + headOff + (size_t)(kn + 16) * HID, HID);
#pragma unroll
      for (int t = 0; t < 4; ++t)
        prefetch_frag(Vt + (headD + t * 16) * SEQ + kn, SEQ);
    }

    // ---- scores: S = Q K^T (two 16-key sub-tiles) ----
    float sc[2][8];
#pragma unroll
    for (int j = 0; j < 2; ++j) {
      v8f c = zero;
      c = wmma32(qa0, kb[j][0], c);
      c = wmma32(qa1, kb[j][1], c);
#pragma unroll
      for (int r = 0; r < 8; ++r) {
        const int gcol = kc + 16 * j + col;
        const int grow = qbase + r + 8 * hi;
        float s = c[r] * scale;
        if (gcol > grow) s = -1.0e9f;  // additive causal mask (== reference)
        sc[j][r] = s;
      }
    }

    // ---- online softmax (xor reductions stay inside each 16-lane half) ----
#pragma unroll
    for (int r = 0; r < 8; ++r) {
      float mx = fmaxf(sc[0][r], sc[1][r]);
      mx = fmaxf(mx, __shfl_xor(mx, 1));
      mx = fmaxf(mx, __shfl_xor(mx, 2));
      mx = fmaxf(mx, __shfl_xor(mx, 4));
      mx = fmaxf(mx, __shfl_xor(mx, 8));
      const float mn    = fmaxf(m[r], mx);
      const float alpha = __expf(m[r] - mn);
      const float p0 = __expf(sc[0][r] - mn);
      const float p1 = __expf(sc[1][r] - mn);
      float rs = p0 + p1;
      rs += __shfl_xor(rs, 1);
      rs += __shfl_xor(rs, 2);
      rs += __shfl_xor(rs, 4);
      rs += __shfl_xor(rs, 8);
      l[r] = l[r] * alpha + rs;
      m[r] = mn;
#pragma unroll
      for (int t = 0; t < 4; ++t) acc[t][r] *= alpha;
      Plds[(r + 8 * hi) * 32 + col]      = (_Float16)p0;
      Plds[(r + 8 * hi) * 32 + 16 + col] = (_Float16)p1;
    }
    __syncthreads();
    v16h pa = load_frag_rowmajor(Plds, 32);     // P as A-fragment (16x32)
#pragma unroll
    for (int t = 0; t < 4; ++t)
      acc[t] = wmma32(pa, vb[t], acc[t]);       // vb already resident
    __syncthreads();
  }

#pragma unroll
  for (int r = 0; r < 8; ++r) {
    const float inv = 1.0f / l[r];
#pragma unroll
    for (int t = 0; t < 4; ++t) {
      const size_t idx = headOff + (size_t)(qbase + r + 8 * hi) * HID + t * 16 + col;
      O[idx] = (_Float16)(acc[t][r] * inv);
    }
  }
}

// ---------------------------------------------------------------------------
// Host orchestration
// ---------------------------------------------------------------------------
extern "C" void kernel_launch(void* const* d_in, const int* in_sizes, int n_in,
                              void* d_out, int out_size, void* d_ws, size_t ws_size,
                              hipStream_t stream) {
  (void)in_sizes; (void)n_in; (void)out_size; (void)ws_size;

  const float* x = (const float*)d_in[0];
  // d_in[1] = attention_mask: exactly causal, folded into the kernel.
  const float* Vh[4] = {(const float*)d_in[2],  (const float*)d_in[6],
                        (const float*)d_in[10], (const float*)d_in[14]};
  const float* S[4]  = {(const float*)d_in[3],  (const float*)d_in[7],
                        (const float*)d_in[11], (const float*)d_in[15]};
  const float* U[4]  = {(const float*)d_in[4],  (const float*)d_in[8],
                        (const float*)d_in[12], (const float*)d_in[16]};
  const float* bb[4] = {(const float*)d_in[5],  (const float*)d_in[9],
                        (const float*)d_in[13], (const float*)d_in[17]};
  const float* W[4]  = {(const float*)d_in[18], (const float*)d_in[20],
                        (const float*)d_in[22], (const float*)d_in[24]};
  const float* bf[4] = {(const float*)d_in[19], (const float*)d_in[21],
                        (const float*)d_in[23], (const float*)d_in[25]};
  float* out = (float*)d_out;

  char* ws = (char*)d_ws;
  size_t off = 0;
  auto alloc = [&](size_t bytes) -> void* {
    void* p = ws + off;
    off = (off + bytes + 255) & ~(size_t)255;
    return p;
  };
  _Float16* x16 = (_Float16*)alloc((size_t)MTOT * HID * 2);
  _Float16 *VhT16[4], *U16[4], *WT16[4];
  for (int i = 0; i < 4; ++i) VhT16[i] = (_Float16*)alloc((size_t)RNK * HID * 2);
  for (int i = 0; i < 4; ++i) U16[i]  = (_Float16*)alloc((size_t)HID * RNK * 2);
  for (int i = 0; i < 4; ++i) WT16[i] = (_Float16*)alloc((size_t)HID * HID * 2);
  _Float16* t16 = (_Float16*)alloc((size_t)MTOT * RNK * 2);
  _Float16* Q16 = (_Float16*)alloc((size_t)MTOT * HID * 2);
  _Float16* K16 = (_Float16*)alloc((size_t)MTOT * HID * 2);
  _Float16* Vt16 = (_Float16*)alloc((size_t)MTOT * HID * 2);  // per-head transposed
  _Float16* C16 = (_Float16*)alloc((size_t)MTOT * HID * 2);

  // ---- prep: f32 -> f16, weights stored as B^T row-major ----
  {
    const int nx = MTOT * HID;
    cvt_f32_f16<<<(nx + 255) / 256, 256, 0, stream>>>(x, x16, nx);
    const int nvr = HID * RNK, nww = HID * HID;
    for (int i = 0; i < 4; ++i) {
      cvtT_f32_f16<<<(nvr + 255) / 256, 256, 0, stream>>>(Vh[i], VhT16[i], HID, RNK);
      cvt_f32_f16<<<(nvr + 255) / 256, 256, 0, stream>>>(U[i], U16[i], nvr); // (U^T)^T = U
      cvtT_f32_f16<<<(nww + 255) / 256, 256, 0, stream>>>(W[i], WT16[i], HID, HID);
    }
  }

  auto gemm = [&](const _Float16* A, int lda, const _Float16* Bt, int ldb,
                  int N, int K, const float* cs, const float* cb,
                  float os, int accum, int mode, float* oF, _Float16* oH) {
    dim3 g(N / 64, MTOT / 32);
    gemm_wmma<<<g, 32, 0, stream>>>(A, lda, Bt, ldb, N, K, cs, cb, os, accum, mode, oF, oH);
  };
  const dim3 attnGrid(SEQ / 16, NHEAD, BS);

  // ---- low-rank SVD path: mixed with weight ALPHA ----
  gemm(x16, HID, VhT16[0], HID, RNK, HID, S[0], nullptr, 1.0f, 0, 0, nullptr, t16);
  gemm(t16, RNK, U16[0],  RNK, HID, RNK, nullptr, bb[0], 1.0f, 0, 0, nullptr, Q16);
  gemm(x16, HID, VhT16[1], HID, RNK, HID, S[1], nullptr, 1.0f, 0, 0, nullptr, t16);
  gemm(t16, RNK, U16[1],  RNK, HID, RNK, nullptr, bb[1], 1.0f, 0, 0, nullptr, K16);
  gemm(x16, HID, VhT16[2], HID, RNK, HID, S[2], nullptr, 1.0f, 0, 0, nullptr, t16);
  gemm(t16, RNK, U16[2],  RNK, HID, RNK, nullptr, bb[2], 1.0f, 0, 1, nullptr, Vt16);
  flash_attn_wmma<<<attnGrid, 32, 0, stream>>>(Q16, K16, Vt16, C16);
  gemm(C16, HID, VhT16[3], HID, RNK, HID, S[3], nullptr, 1.0f, 0, 0, nullptr, t16);
  gemm(t16, RNK, U16[3],  RNK, HID, RNK, nullptr, bb[3], ALPHA_MIX, 0, 0, out, nullptr);

  // ---- full-rank fallback path: accumulated with weight (1-ALPHA) ----
  gemm(x16, HID, WT16[0], HID, HID, HID, nullptr, bf[0], 1.0f, 0, 0, nullptr, Q16);
  gemm(x16, HID, WT16[1], HID, HID, HID, nullptr, bf[1], 1.0f, 0, 0, nullptr, K16);
  gemm(x16, HID, WT16[2], HID, HID, HID, nullptr, bf[2], 1.0f, 0, 1, nullptr, Vt16);
  flash_attn_wmma<<<attnGrid, 32, 0, stream>>>(Q16, K16, Vt16, C16);
  gemm(C16, HID, WT16[3], HID, HID, HID, nullptr, bf[3], 1.0f - ALPHA_MIX, 1, 0, out, nullptr);
}
